// BitwiseLinear_20126216749915
// MI455X (gfx1250) — matrix operand order repaired and checked by reading, not compiled, not run on hardware
//
#include <hip/hip_runtime.h>
#include <hip/hip_bf16.h>
#include <math.h>

#define TOKENS  32768
#define IN_DIM  8192
#define OUT_DIM 256

typedef __bf16 v16bf __attribute__((ext_vector_type(16)));
typedef __bf16 v8bf  __attribute__((ext_vector_type(8)));
typedef __bf16 v4bf  __attribute__((ext_vector_type(4)));
typedef float  v16f  __attribute__((ext_vector_type(16)));
typedef float  v8f   __attribute__((ext_vector_type(8)));
typedef float  v4f   __attribute__((ext_vector_type(4)));

// ---------------------------------------------------------------------------
// Stage 1: wb[n][k] = bf16(tanh(W[n][k])).  2M elements, 4 per thread.
// ---------------------------------------------------------------------------
__global__ __launch_bounds__(256)
void tanh_w_to_bf16(const v4f* __restrict__ w, v4bf* __restrict__ wb) {
    int i = blockIdx.x * 256 + threadIdx.x;          // 524288 vec4 elements
    v4f v = w[i];
    v4f t;
    t[0] = tanhf(v[0]);
    t[1] = tanhf(v[1]);
    t[2] = tanhf(v[2]);
    t[3] = tanhf(v[3]);
    wb[i] = __builtin_convertvector(t, v4bf);
}

// ---------------------------------------------------------------------------
// Stage 2: y = x @ wb^T via v_wmma_f32_16x16x32_bf16.
// Block = 256 threads = 8 waves: 4 M-groups x 2 N-halves.
// Wave tile: 16 M x 128 N (8 WMMA tiles). Block tile: 64 M x 256 N.
// Grid: TOKENS/64 = 512 blocks. Each x row streamed from HBM exactly once.
// ---------------------------------------------------------------------------
__global__ __launch_bounds__(256)
void gemm_bf16_wmma(const float* __restrict__ x,
                    const __bf16* __restrict__ wb,
                    float* __restrict__ out) {
    const int lane = threadIdx.x & 31;
    const int wave = threadIdx.x >> 5;
    const int wm   = wave >> 1;                  // 0..3 : M sub-block
    const int wn   = wave &  1;                  // 0..1 : N half
    const int row0 = blockIdx.x * 64 + wm * 16;  // token row base
    const int col0 = wn * 128;                   // output-channel base

    const int hi16 = lane >> 4;                  // 0 | 1
    const int l15  = lane & 15;

    // A layout (16-bit 16x32): lane<16 -> row=l15, K chunks {0..7,16..23};
    //                          lane>=16 -> K chunks {8..15,24..31}.
    const float* xp = x + (size_t)(row0 + l15) * IN_DIM + hi16 * 8;

    // B layout (16-bit 32x16): lanes 0-15 -> N=l15, K chunks {0..7,16..23};
    //                          lanes 16-31 -> K chunks {8..15,24..31}.  W row == B column.
    const __bf16* wp = wb + (size_t)(col0 + l15) * IN_DIM + hi16 * 8;

    v8f acc[8];
#pragma unroll
    for (int t = 0; t < 8; ++t)
        acc[t] = (v8f){0.f, 0.f, 0.f, 0.f, 0.f, 0.f, 0.f, 0.f};

    for (int k = 0; k < IN_DIM; k += 32) {
        // ---- A tile: 16 f32 per lane -> packed cvt to bf16 ----
        v4f f0 = *(const v4f*)(xp + k);
        v4f f1 = *(const v4f*)(xp + k + 4);
        v4f f2 = *(const v4f*)(xp + k + 16);
        v4f f3 = *(const v4f*)(xp + k + 20);
        // prefetch next K-step of this x row into the cache hierarchy
        __builtin_prefetch(xp + k + 32, 0, 1);

        v8f lo = __builtin_shufflevector(f0, f1, 0, 1, 2, 3, 4, 5, 6, 7);
        v8f hi = __builtin_shufflevector(f2, f3, 0, 1, 2, 3, 4, 5, 6, 7);
        v16f af = __builtin_shufflevector(lo, hi, 0, 1, 2, 3, 4, 5, 6, 7,
                                                  8, 9, 10, 11, 12, 13, 14, 15);
        v16bf a = __builtin_convertvector(af, v16bf);

        // ---- 8 B tiles (bf16 straight from L2-resident wb) + 8 WMMAs ----
#pragma unroll
        for (int t = 0; t < 8; ++t) {
            const __bf16* bp = wp + (size_t)t * 16 * IN_DIM + k;
            v8bf b0 = *(const v8bf*)(bp);
            v8bf b1 = *(const v8bf*)(bp + 16);
            v16bf b = __builtin_shufflevector(b0, b1, 0, 1, 2, 3, 4, 5, 6, 7,
                                                      8, 9, 10, 11, 12, 13, 14, 15);
            acc[t] = __builtin_amdgcn_wmma_f32_16x16x32_bf16(
                         /*neg_a=*/false, a, /*neg_b=*/false, b,
                         /*c_mod=*/(short)0, acc[t],
                         /*reuse_a=*/false, /*reuse_b=*/false);
        }
    }

    // ---- C store: VGPR j -> M = j + 8*hi16, N = l15 (per 16x16 tile) ----
    const int mbase = row0 + hi16 * 8;
#pragma unroll
    for (int t = 0; t < 8; ++t) {
        float* op = out + (size_t)mbase * OUT_DIM + col0 + t * 16 + l15;
#pragma unroll
        for (int j = 0; j < 8; ++j)
            op[(size_t)j * OUT_DIM] = acc[t][j];
    }
}

// ---------------------------------------------------------------------------
extern "C" void kernel_launch(void* const* d_in, const int* in_sizes, int n_in,
                              void* d_out, int out_size, void* d_ws, size_t ws_size,
                              hipStream_t stream) {
    const float* x = (const float*)d_in[0];   // [TOKENS, IN_DIM] f32
    const float* w = (const float*)d_in[1];   // [OUT_DIM, IN_DIM] f32
    float* out = (float*)d_out;               // [TOKENS, OUT_DIM] f32
    __bf16* wb = (__bf16*)d_ws;               // [OUT_DIM, IN_DIM] bf16 (4 MiB)

    (void)in_sizes; (void)n_in; (void)out_size; (void)ws_size;

    // Stage 1: tanh + bf16 pack of the weight (2M elems, 4/thread)
    tanh_w_to_bf16<<<(OUT_DIM * IN_DIM / 4) / 256, 256, 0, stream>>>(
        (const v4f*)w, (v4bf*)wb);

    // Stage 2: WMMA GEMM, 512 blocks x 256 threads
    gemm_bf16_wmma<<<TOKENS / 64, 256, 0, stream>>>(x, wb, out);
}
